// HGCN_34600256537155
// MI455X (gfx1250) — matrix operand order repaired and checked by reading, not compiled, not run on hardware
//
#include <hip/hip_runtime.h>
#include <hip/hip_bf16.h>
#include <math.h>

// ---------------------------------------------------------------------------
// HGCN for MI455X (gfx1250, wave32, WMMA).
//   * conv reordered:  out = rin ⊙ scatter_add( (rout ⊙ h) @ W ) + b
//   * all GEMMs on v_wmma_f32_16x16x32_bf16; weights pre-packed to per-lane B
//     fragment layout; each wave computes a 16 x (NB*16) strip so one A
//     fragment (load + deg-scale + bf16 convert) feeds NB WMMAs.
// ---------------------------------------------------------------------------

typedef __attribute__((ext_vector_type(16))) __bf16 v16bf;
typedef __attribute__((ext_vector_type(8)))  float  v8f;

#define N_SAM 10000
#define N_GEN 20000
#define SLOPE 0.25f

// ---------------------------------------------------------------- utilities
__global__ void zero_kernel(float* __restrict__ p, int n) {
    int i = blockIdx.x * 256 + threadIdx.x;
    if (i < n) p[i] = 0.0f;
}

__global__ void deg_kernel(const int* __restrict__ s1, const int* __restrict__ d1,
                           const int* __restrict__ s2, const int* __restrict__ d2,
                           float* __restrict__ ds1, float* __restrict__ dd1,
                           float* __restrict__ ds2, float* __restrict__ dd2, int nE) {
    int e = blockIdx.x * 256 + threadIdx.x;
    if (e >= nE) return;
    atomicAdd(&ds1[s1[e]], 1.0f);
    atomicAdd(&dd1[d1[e]], 1.0f);
    atomicAdd(&ds2[s2[e]], 1.0f);
    atomicAdd(&dd2[d2[e]], 1.0f);
}

__global__ void rsqrt_kernel(float* __restrict__ d, int n) {
    int i = blockIdx.x * 256 + threadIdx.x;
    if (i >= n) return;
    float v = d[i];
    d[i] = rsqrtf(v < 1.0f ? 1.0f : v);
}

// ------------------------------------------------- weight pack (B fragments)
// packed[nt][kt][lane][j] : lane 0-15 -> col n = nt*16 + (lane&15), K = kt*32 + (j&7) + 16*(j>>3)
//                           lane16-31 -> same col,                K = kt*32 + 8 + (j&7) + 16*(j>>3)
// K tail zero-padded so the GEMM hot loop needs no B predication.
__global__ void pack_w(const float* __restrict__ W, __bf16* __restrict__ out,
                       int K, int N, int KT) {
    int idx = blockIdx.x * 256 + threadIdx.x;
    int total = (N >> 4) * KT * 512;
    if (idx >= total) return;
    int j    = idx & 15;
    int lane = (idx >> 4) & 31;
    int kt   = (idx >> 9) % KT;
    int nt   = idx / (KT * 512);
    int n = (nt << 4) + (lane & 15);
    int k = kt * 32 + ((lane >> 4) << 3) + (j & 7) + ((j >> 3) << 4);
    out[idx] = (k < K) ? (__bf16)W[(size_t)k * N + n] : (__bf16)0.0f;
}

// ------------------------------------------------------------- WMMA GEMM
// C[M,N] = (scale ? diag(scale) : I) * A[M,K] @ Wpacked + (bias ? bias : 0)
// one wave per 16 x (NB*16) strip; blockDim = (32, 8).
template <int NB>
__global__ __launch_bounds__(256)
void wmma_gemm(const float* __restrict__ A, const __bf16* __restrict__ Bp,
               const float* __restrict__ scale, const float* __restrict__ bias,
               float* __restrict__ C, int M, int N, int K, int KT) {
    const int lane   = threadIdx.x;
    const int wave   = threadIdx.y;
    const int groups = (N >> 4) / NB;
    const int tile   = blockIdx.x * 8 + wave;
    if (tile >= (M >> 4) * groups) return;     // wave-uniform: EXEC stays all-1s
    const int mt = tile / groups;
    const int g  = tile - mt * groups;
    const int hi  = lane >> 4;                 // half-wave id
    const int row = (mt << 4) + (lane & 15);
    const float s = scale ? scale[row] : 1.0f;
    const float* __restrict__ arow = A + (size_t)row * K;
    const v16bf* __restrict__ bp =
        (const v16bf*)Bp + (size_t)g * NB * KT * 32 + lane;

    v8f acc[NB];
#pragma unroll
    for (int nb = 0; nb < NB; ++nb) acc[nb] = (v8f){};

    for (int kt = 0; kt < KT; ++kt) {
        const int k0 = kt * 32 + hi * 8;   // A: j 0-7 -> k0+j ; j 8-15 -> k0+16+(j-8)
        v16bf af;
        if ((kt + 1) * 32 <= K) {          // uniform branch
            float4 a0 = *(const float4*)(arow + k0);
            float4 a1 = *(const float4*)(arow + k0 + 4);
            float4 a2 = *(const float4*)(arow + k0 + 16);
            float4 a3 = *(const float4*)(arow + k0 + 20);
            af[0]  = (__bf16)(a0.x * s); af[1]  = (__bf16)(a0.y * s);
            af[2]  = (__bf16)(a0.z * s); af[3]  = (__bf16)(a0.w * s);
            af[4]  = (__bf16)(a1.x * s); af[5]  = (__bf16)(a1.y * s);
            af[6]  = (__bf16)(a1.z * s); af[7]  = (__bf16)(a1.w * s);
            af[8]  = (__bf16)(a2.x * s); af[9]  = (__bf16)(a2.y * s);
            af[10] = (__bf16)(a2.z * s); af[11] = (__bf16)(a2.w * s);
            af[12] = (__bf16)(a3.x * s); af[13] = (__bf16)(a3.y * s);
            af[14] = (__bf16)(a3.z * s); af[15] = (__bf16)(a3.w * s);
        } else {                           // K%32 tail: clamped loads, no EXEC games
#pragma unroll
            for (int j = 0; j < 16; ++j) {
                int k  = k0 + j + ((j >= 8) ? 8 : 0);
                int kc = (k < K) ? k : (K - 1);
                float v = arow[kc] * s;
                af[j] = (__bf16)((k < K) ? v : 0.0f);
            }
        }
#pragma unroll
        for (int nb = 0; nb < NB; ++nb) {
            v16bf bfrag = bp[((size_t)nb * KT + kt) * 32];
            acc[nb] = __builtin_amdgcn_wmma_f32_16x16x32_bf16(
                false, af, false, bfrag, (short)0, acc[nb], false, false);
        }
    }
    // C/D layout: VGPR i -> row (i + 8*hi), col = lane&15 within tile
#pragma unroll
    for (int nb = 0; nb < NB; ++nb) {
        const int col = ((g * NB + nb) << 4) + (lane & 15);
        const float b = bias ? bias[col] : 0.0f;
        float* crow = C + ((size_t)((mt << 4) + hi * 8)) * N + col;
#pragma unroll
        for (int i = 0; i < 8; ++i) crow[(size_t)i * N] = acc[nb][i] + b;
    }
}

// ----------------------------------------------------------- edge scatter
// acc[dst[e]][:] += t[src[e]][:]  ; one thread per (edge, float4 chunk)
__global__ void scatter_kernel(const float* __restrict__ t, const int* __restrict__ src,
                               const int* __restrict__ dst, float* __restrict__ acc,
                               int d4, int nE) {
    int tid = blockIdx.x * 256 + threadIdx.x;
    if (tid >= nE * d4) return;
    int e = tid / d4;
    int c = tid - e * d4;
    int d = d4 << 2;
    float4 v = ((const float4*)(t + (size_t)src[e] * d))[c];
    float* a = acc + (size_t)dst[e] * d + (c << 2);
    atomicAdd(a + 0, v.x);
    atomicAdd(a + 1, v.y);
    atomicAdd(a + 2, v.z);
    atomicAdd(a + 3, v.w);
}

// ------------------------------------------------- epilogue: rin ⊙ acc + b, leaky
__global__ void finalize_kernel(const float* __restrict__ acc, const float* __restrict__ rin,
                                const float* __restrict__ bias, float* __restrict__ out,
                                int n, int d) {
    int tid = blockIdx.x * 256 + threadIdx.x;
    if (tid >= n * d) return;
    int i = tid / d;
    int j = tid - i * d;
    float v = acc[tid] * rin[i] + bias[j];
    out[tid] = v > 0.0f ? v : SLOPE * v;
}

// ---------------------------------------------------------------------------
extern "C" void kernel_launch(void* const* d_in, const int* in_sizes, int n_in,
                              void* d_out, int out_size, void* d_ws, size_t ws_size,
                              hipStream_t stream) {
    const float* sam_feat = (const float*)d_in[0];
    const float* gen_feat = (const float*)d_in[1];
    const int* src_sg = (const int*)d_in[2];
    const int* dst_sg = (const int*)d_in[3];
    const int* src_gs = (const int*)d_in[4];
    const int* dst_gs = (const int*)d_in[5];
    const float* l1W = (const float*)d_in[6];
    const float* l1b = (const float*)d_in[7];
    const float* l2W = (const float*)d_in[8];
    const float* l2b = (const float*)d_in[9];
    const float* Wmat[3][2], *bvec[3][2];     // [layer][0]=sg, [1]=gs
    for (int l = 0; l < 3; ++l) {
        Wmat[l][0] = (const float*)d_in[10 + l * 4 + 0];
        bvec[l][0] = (const float*)d_in[10 + l * 4 + 1];
        Wmat[l][1] = (const float*)d_in[10 + l * 4 + 2];
        bvec[l][1] = (const float*)d_in[10 + l * 4 + 3];
    }
    const int nE = in_sizes[2];

    // ---- workspace carve-up (~63.4 MB) ----
    float* ws = (float*)d_ws;
    const size_t HS = (size_t)N_SAM * 256;
    const size_t HG = (size_t)N_GEN * 256;
    float* h_sam = ws;                     // doubles as acc_sam
    float* t_sam = ws + HS;
    float* h_gen = ws + 2 * HS;            // doubles as acc_gen
    float* t_gen = ws + 2 * HS + HG;
    float* r_src_sg = ws + 2 * HS + 2 * HG;          // N_SAM
    float* r_dst_sg = r_src_sg + N_SAM;              // N_GEN
    float* r_src_gs = r_dst_sg + N_GEN;              // N_GEN
    float* r_dst_gs = r_src_gs + N_GEN;              // N_SAM
    __bf16* wpack = (__bf16*)(r_dst_gs + N_SAM);

    // ---- pack all weights to bf16 B-fragments ----
    size_t woff = 0;
    auto pack = [&](const float* W, int K, int N) -> __bf16* {
        int KT = (K + 31) / 32;
        size_t total = (size_t)(N >> 4) * KT * 512;
        __bf16* p = wpack + woff;
        woff += total;
        pack_w<<<(unsigned)((total + 255) / 256), 256, 0, stream>>>(W, p, K, N, KT);
        return p;
    };
    __bf16* p_l1 = pack(l1W, 2000, 256);
    __bf16* p_l2 = pack(l2W, 500, 256);
    const int dims[4] = {256, 256, 128, 64};
    __bf16* pW[3][2];
    for (int l = 0; l < 3; ++l) {
        pW[l][0] = pack(Wmat[l][0], dims[l], dims[l + 1]);
        pW[l][1] = pack(Wmat[l][1], dims[l], dims[l + 1]);
    }

    // ---- degrees -> rsqrt(clip(deg,1)), shared by all layers ----
    const int ndeg = 2 * N_SAM + 2 * N_GEN;
    zero_kernel<<<(ndeg + 255) / 256, 256, 0, stream>>>(r_src_sg, ndeg);
    deg_kernel<<<(nE + 255) / 256, 256, 0, stream>>>(src_sg, dst_sg, src_gs, dst_gs,
                                                     r_src_sg, r_dst_sg, r_src_gs, r_dst_gs, nE);
    rsqrt_kernel<<<(ndeg + 255) / 256, 256, 0, stream>>>(r_src_sg, ndeg);

    auto gemm = [&](const float* A, const __bf16* Bp, const float* sc, const float* bi,
                    float* C, int M, int N, int K) {
        int KT = (K + 31) / 32;
        dim3 blk(32, 8);
        if ((N & 127) == 0) {  // N = 256 or 128 -> NB=8
            int tiles = (M >> 4) * ((N >> 4) / 8);
            wmma_gemm<8><<<(tiles + 7) / 8, blk, 0, stream>>>(A, Bp, sc, bi, C, M, N, K, KT);
        } else {               // N = 64 -> NB=4
            int tiles = (M >> 4) * ((N >> 4) / 4);
            wmma_gemm<4><<<(tiles + 7) / 8, blk, 0, stream>>>(A, Bp, sc, bi, C, M, N, K, KT);
        }
    };

    // ---- input projections ----
    gemm(sam_feat, p_l1, nullptr, l1b, h_sam, N_SAM, 256, 2000);
    gemm(gen_feat, p_l2, nullptr, l2b, h_gen, N_GEN, 256, 500);

    // ---- 3 hetero conv layers ----
    float* out_sam = (float*)d_out;                        // [10000,64]
    float* out_gen = (float*)d_out + (size_t)N_SAM * 64;   // [20000,64]
    for (int l = 0; l < 3; ++l) {
        const int di = dims[l], dn = dims[l + 1];
        // t = (rout ⊙ h) @ W   (W applied before aggregation)
        gemm(h_sam, pW[l][0], r_src_sg, nullptr, t_sam, N_SAM, dn, di);
        gemm(h_gen, pW[l][1], r_src_gs, nullptr, t_gen, N_GEN, dn, di);
        // h buffers are dead now -> reuse as accumulators
        zero_kernel<<<(N_SAM * dn + 255) / 256, 256, 0, stream>>>(h_sam, N_SAM * dn);
        zero_kernel<<<(N_GEN * dn + 255) / 256, 256, 0, stream>>>(h_gen, N_GEN * dn);
        const int d4 = dn >> 2;
        scatter_kernel<<<(nE * d4 + 255) / 256, 256, 0, stream>>>(t_sam, src_sg, dst_sg, h_gen, d4, nE);
        scatter_kernel<<<(nE * d4 + 255) / 256, 256, 0, stream>>>(t_gen, src_gs, dst_gs, h_sam, d4, nE);
        // epilogue: deg_in^{-1/2} ⊙ acc + b, leaky relu (in place except last layer)
        float* osam = (l == 2) ? out_sam : h_sam;
        float* ogen = (l == 2) ? out_gen : h_gen;
        finalize_kernel<<<(N_SAM * dn + 255) / 256, 256, 0, stream>>>(h_sam, r_dst_gs, bvec[l][1], osam, N_SAM, dn);
        finalize_kernel<<<(N_GEN * dn + 255) / 256, 256, 0, stream>>>(h_gen, r_dst_sg, bvec[l][0], ogen, N_GEN, dn);
    }
}
